// CCPM_5970004542310
// MI455X (gfx1250) — matrix operand order repaired and checked by reading, not compiled, not run on hardware
//
#include <hip/hip_runtime.h>
#include <hip/hip_bf16.h>

// CCPM fused kernel for gfx1250 (MI455X).
// One sample per wave32; convs expressed as 32x32 matrices (kernel spans the
// full embedding axis with SAME padding) and executed with
// v_wmma_f32_16x16x32_f16 (f16 in / f32 accumulate).

typedef __attribute__((ext_vector_type(16))) _Float16 v16h;
typedef __attribute__((ext_vector_type(8)))  _Float16 v8h;
typedef __attribute__((ext_vector_type(8)))  float    v8f;

#define NFIELD 16
#define EMBED  32
#define VOCAB  100000
#define BATCHN 4096
#define WAVES_PER_BLK 4

union V16U { v16h v; v8h h[2]; };

__global__ __launch_bounds__(128) void ccpm_fused(
    const int*   __restrict__ idx,
    const float* __restrict__ w0,
    const float* __restrict__ b0,
    const float* __restrict__ f1,
    const float* __restrict__ f2,
    const float* __restrict__ w1,
    const float* __restrict__ b1,
    float*       __restrict__ out)
{
  // Filters rearranged + zero padded so A-fragment gathers are branch-free:
  // f1r[kw][oc][m], m = kh+16, zeros outside kh in [0,32)
  __shared__ __align__(16) _Float16 f1r[7][2][64];
  __shared__ __align__(16) _Float16 f2r[5][2][2][64];
  __shared__ float w1s[192];
  // Per-wave activation tiles, stored TRANSPOSED [col][h] so B-fragments are
  // contiguous 16-half runs per lane (ds_load_b128 x2).
  __shared__ __align__(16) _Float16 Xp[WAVES_PER_BLK][24][32];     // conv1 input, cols = w+kw
  __shared__ __align__(16) _Float16 Yp[WAVES_PER_BLK][2][24][32];  // conv2 input per ic
  // Per-wave f32 scratch: [0..1023] conv outputs, [1024..1215] z vector
  __shared__ __align__(16) float ob[WAVES_PER_BLK][1024 + 192];

  const int tid  = threadIdx.x;
  const int wv   = tid >> 5;
  const int lane = tid & 31;
  const int nlo  = lane & 15;   // N index (column) of WMMA fragments
  const int hi   = lane >> 4;   // lane half selects K range
  const int b    = blockIdx.x * WAVES_PER_BLK + wv;

  // ---- cooperative filter prep ----
  for (int i = tid; i < 7 * 2 * 64; i += 128) {
    int m = i & 63, t = i >> 6;             // t = kw*2 + oc
    int kh = m - 16;
    float v = (kh >= 0 && kh < 32) ? f1[kh * 14 + t] : 0.0f;
    ((_Float16*)f1r)[i] = (_Float16)v;
  }
  for (int i = tid; i < 5 * 2 * 2 * 64; i += 128) {
    int m = i & 63, t = i >> 6;             // t = (kw*2 + ic)*2 + oc
    int kh = m - 16;
    float v = (kh >= 0 && kh < 32) ? f2[kh * 20 + t] : 0.0f;
    ((_Float16*)f2r)[i] = (_Float16)v;
  }
  for (int i = tid; i < 192; i += 128) w1s[i] = w1[i];

  // zero padded activation tiles (vectorized)
  {
    uint4 zz = make_uint4(0u, 0u, 0u, 0u);
    uint4* px = (uint4*)&Xp[wv][0][0];              // 1536 B = 96 uint4
    for (int i = lane; i < 96; i += 32) px[i] = zz;
    uint4* py = (uint4*)&Yp[wv][0][0][0];           // 3072 B = 192 uint4
    for (int i = lane; i < 192; i += 32) py[i] = zz;
  }
  __syncthreads();

  // ---- embedding gather + tanh, lane = embedding dim e ----
  for (int f = 0; f < NFIELD; ++f) {
    int iv = idx[b * NFIELD + f];
    float v = w0[((long long)f * VOCAB + iv) * EMBED + lane] + b0[f * EMBED + lane];
    Xp[wv][f + 3][lane] = (_Float16)tanhf(v);       // col = w'+3 (pad_lo = 3)
  }
  __syncthreads();

  // ---- conv1 as WMMA: out1[h,w,oc] = sum_kw A1[oc,kw](32x32) @ X[:, w+kw-3] ----
  v8f acc[2][2] = {};   // [oc][mtile]
#pragma unroll
  for (int kw = 0; kw < 7; ++kw) {
    V16U bf;
    const _Float16* xc = &Xp[wv][nlo + kw][hi * 16];
    bf.h[0] = *(const v8h*)(xc);
    bf.h[1] = *(const v8h*)(xc + 8);
#pragma unroll
    for (int oc = 0; oc < 2; ++oc) {
#pragma unroll
      for (int mt = 0; mt < 2; ++mt) {
        // A layout: M = mt*16 + nlo, K(j) = j + hi*16
        // kh = K - M + 15  ->  padded index = K - M + 31
        int base = 31 - (mt * 16 + nlo) + hi * 16;
        V16U af;
#pragma unroll
        for (int j = 0; j < 16; ++j)
          af.v[j] = f1r[kw][oc][base + j];
        acc[oc][mt] = __builtin_amdgcn_wmma_f32_16x16x32_f16(
            false, af.v, false, bf.v, (short)0, acc[oc][mt], false, false);
      }
    }
  }

  // spill conv1 result: C/D layout M = r + hi*8 (+16*mt), N = nlo
  float* o1 = ob[wv];
#pragma unroll
  for (int oc = 0; oc < 2; ++oc)
#pragma unroll
    for (int mt = 0; mt < 2; ++mt)
#pragma unroll
      for (int r = 0; r < 8; ++r)
        o1[(mt * 16 + hi * 8 + r) * 32 + nlo * 2 + oc] = acc[oc][mt][r];
  __syncthreads();

  // ---- k-max pool 1: top-8 over field axis, sorted descending (== top_k) ----
#pragma unroll
  for (int s = 0; s < 2; ++s) {
    const int h = lane, ic = s;     // 64 rows = 32 lanes x 2
    float t[8];
#pragma unroll
    for (int i = 0; i < 8; ++i) t[i] = -3.402823466e38f;
    for (int w = 0; w < 16; ++w) {
      float v = o1[h * 32 + w * 2 + ic];
#pragma unroll
      for (int i = 0; i < 8; ++i)
        if (v > t[i]) { float tmp = t[i]; t[i] = v; v = tmp; }
    }
#pragma unroll
    for (int k = 0; k < 8; ++k)
      Yp[wv][ic][k + 2][h] = (_Float16)t[k];        // col = k+2 (pad_lo = 2)
  }
  __syncthreads();

  // ---- conv2 as WMMA: out2[h,k,oc] = sum_{kw,ic} A2 @ Y[ic][:, k+kw-2] ----
  v8f acc2[2][2] = {};
#pragma unroll
  for (int kw = 0; kw < 5; ++kw) {
#pragma unroll
    for (int ic = 0; ic < 2; ++ic) {
      V16U bf;
      const _Float16* yc = &Yp[wv][ic][nlo + kw][hi * 16];
      bf.h[0] = *(const v8h*)(yc);
      bf.h[1] = *(const v8h*)(yc + 8);
#pragma unroll
      for (int oc = 0; oc < 2; ++oc) {
#pragma unroll
        for (int mt = 0; mt < 2; ++mt) {
          int base = 31 - (mt * 16 + nlo) + hi * 16;
          V16U af;
#pragma unroll
          for (int j = 0; j < 16; ++j)
            af.v[j] = f2r[kw][ic][oc][base + j];
          acc2[oc][mt] = __builtin_amdgcn_wmma_f32_16x16x32_f16(
              false, af.v, false, bf.v, (short)0, acc2[oc][mt], false, false);
        }
      }
    }
  }

  // spill conv2 (only k = 0..7 columns are real outputs)
  if (nlo < 8) {
#pragma unroll
    for (int oc = 0; oc < 2; ++oc)
#pragma unroll
      for (int mt = 0; mt < 2; ++mt)
#pragma unroll
        for (int r = 0; r < 8; ++r)
          o1[(mt * 16 + hi * 8 + r) * 16 + nlo * 2 + oc] = acc2[oc][mt][r];
  }
  __syncthreads();

  // ---- k-max pool 2: top-3 descending; z layout = h*6 + kk*2 + oc ----
  float* z = ob[wv] + 1024;
#pragma unroll
  for (int s = 0; s < 2; ++s) {
    const int h = lane, oc = s;
    float t[3];
#pragma unroll
    for (int i = 0; i < 3; ++i) t[i] = -3.402823466e38f;
    for (int k = 0; k < 8; ++k) {
      float v = o1[h * 16 + k * 2 + oc];
#pragma unroll
      for (int i = 0; i < 3; ++i)
        if (v > t[i]) { float tmp = t[i]; t[i] = v; v = tmp; }
    }
#pragma unroll
    for (int kk = 0; kk < 3; ++kk)
      z[h * 6 + kk * 2 + oc] = t[kk];
  }
  __syncthreads();

  // ---- tanh -> dot(w1) -> +b1 -> sigmoid (wave32 shuffle reduction) ----
  float sum = 0.0f;
#pragma unroll
  for (int i = 0; i < 6; ++i) {
    int j = lane * 6 + i;
    sum += tanhf(z[j]) * w1s[j];
  }
#pragma unroll
  for (int off = 16; off > 0; off >>= 1)
    sum += __shfl_xor(sum, off, 32);
  if (lane == 0)
    out[b] = 1.0f / (1.0f + expf(-(sum + b1[0])));
}

extern "C" void kernel_launch(void* const* d_in, const int* in_sizes, int n_in,
                              void* d_out, int out_size, void* d_ws, size_t ws_size,
                              hipStream_t stream) {
  const int*   idx = (const int*)  d_in[0];
  const float* w0  = (const float*)d_in[1];
  const float* b0  = (const float*)d_in[2];
  const float* f1  = (const float*)d_in[3];
  const float* f2  = (const float*)d_in[4];
  const float* w1  = (const float*)d_in[5];
  const float* b1  = (const float*)d_in[6];
  float* out = (float*)d_out;

  ccpm_fused<<<BATCHN / WAVES_PER_BLK, 128, 0, stream>>>(
      idx, w0, b0, f1, f2, w1, b1, out);
}
